// GAT_53824530153501
// MI455X (gfx1250) — matrix operand order repaired and checked by reading, not compiled, not run on hardware
//
#include <hip/hip_runtime.h>

// ---------------------------------------------------------------------------
// GAT layer for MI455X (gfx1250, wave32), fused softmax-AV with an analytic
// row-max bound:  s_ij = lrelu(s_src_i + s_dst_j) is monotone in s_dst_j, so
// M_i = lrelu(s_src_i + max_j s_dst_j) >= all valid scores of row i. This
// removes all online-softmax rescaling/reductions from the streaming loop.
// adj (256MB) streamed exactly once with non-temporal loads (preserves 192MB
// L2 for the hot x'^T tiles); attention matrix never materialized.
// Both GEMMs use v_wmma_f32_16x16x32_bf16 with f32 accumulation.
// ---------------------------------------------------------------------------

typedef __attribute__((ext_vector_type(16))) __bf16 v16bf;
typedef __attribute__((ext_vector_type(8)))  float  v8f;

#define N_NODES 8192
#define F_IN    256
#define F_OUT   128
#define TJ      64          // column chunk = 2 x WMMA K
#define NEG_BIG (-1.0e9f)

union ABu { v16bf v; uint4 q[2]; };

__device__ __forceinline__ unsigned short f2bf(float f) {
    unsigned int u = __float_as_uint(f);
    u += 0x7FFFu + ((u >> 16) & 1u);      // round-to-nearest-even
    return (unsigned short)(u >> 16);
}

__device__ __forceinline__ float lrelu(float s) {
    return s >= 0.f ? s : 0.2f * s;
}

// masked score -> probability against fixed bound M (exp(-1e9) underflows to 0)
__device__ __forceinline__ float prob(float raw, float a, int diag, float M) {
    const float s = ((a > 0.f) || diag) ? lrelu(raw) : NEG_BIG;
    return __expf(s - M);
}

// --------------------------- kernel 0: prep --------------------------------
__global__ void __launch_bounds__(256) gat_prep(
    const float* __restrict__ x, const float* __restrict__ w,
    unsigned short* __restrict__ xbf, unsigned short* __restrict__ wT,
    float* __restrict__ s_src, float* __restrict__ s_dst)
{
    int t = blockIdx.x * 256 + threadIdx.x;
    if (t < N_NODES * F_IN) xbf[t] = f2bf(x[t]);
    if (t < F_IN * F_OUT) {
        int k = t >> 7, n = t & 127;              // weight[k][n], row-major
        wT[n * F_IN + k] = f2bf(w[k * F_OUT + n]);
    }
    if (t < N_NODES) { s_src[t] = 0.f; s_dst[t] = 0.f; }
}

// ------------------------ kernel 1: x_prime + s ----------------------------
__global__ void __launch_bounds__(256) gat_xprime(
    const unsigned short* __restrict__ xbf,
    const unsigned short* __restrict__ wT,
    const float* __restrict__ bias,
    const float* __restrict__ phi,
    unsigned short* __restrict__ xpT,
    float* __restrict__ s_src, float* __restrict__ s_dst)
{
    const int lane = threadIdx.x & 31;
    const int w    = threadIdx.x >> 5;
    const int half = lane >> 4;
    const int l15  = lane & 15;
    const int i0   = blockIdx.x * 16;
    const int n0   = w * 16;

    const int arow = i0 + l15;
    const int bcol = n0 + l15;

    v8f c = {};
    #pragma unroll
    for (int kc = 0; kc < F_IN / 32; ++kc) {
        const int k0 = kc * 32;
        ABu a, b;
        const unsigned short* ap = xbf + (size_t)arow * F_IN + k0 + half * 16;
        a.q[0] = *(const uint4*)(ap);
        a.q[1] = *(const uint4*)(ap + 8);
        const unsigned short* bp = wT + (size_t)bcol * F_IN + k0 + half * 16;
        b.q[0] = ((const uint4*)bp)[0];
        b.q[1] = ((const uint4*)bp)[1];
        c = __builtin_amdgcn_wmma_f32_16x16x32_bf16(
                false, a.v, false, b.v, (short)0, c, false, false);
    }

    const float bv   = bias[bcol];
    const float ph_s = phi[bcol];
    const float ph_d = phi[F_OUT + bcol];
    #pragma unroll
    for (int v = 0; v < 8; ++v) {
        const float val = c[v] + bv;
        const int   row = i0 + v + half * 8;
        xpT[(size_t)bcol * N_NODES + row] = f2bf(val);
        float ps = val * ph_s;
        float pd = val * ph_d;
        #pragma unroll
        for (int off = 1; off < 16; off <<= 1) {
            ps += __shfl_xor(ps, off, 32);
            pd += __shfl_xor(pd, off, 32);
        }
        if (l15 == 0) {
            atomicAdd(&s_src[row], ps);
            atomicAdd(&s_dst[row], pd);
        }
    }
}

// -------------------- kernel 1b: global max of s_dst -----------------------
__global__ void __launch_bounds__(256) gat_gmax(
    const float* __restrict__ s_dst, float* __restrict__ gmax)
{
    __shared__ float red[256];
    float m = -3.0e38f;
    for (int k = threadIdx.x; k < N_NODES; k += 256)
        m = fmaxf(m, s_dst[k]);
    red[threadIdx.x] = m;
    __syncthreads();
    #pragma unroll
    for (int s = 128; s >= 1; s >>= 1) {
        if (threadIdx.x < s)
            red[threadIdx.x] = fmaxf(red[threadIdx.x], red[threadIdx.x + s]);
        __syncthreads();
    }
    if (threadIdx.x == 0) gmax[0] = red[0];
}

// ---------------- kernel 2: fused mask/softmax/AV --------------------------
// Block = 16 rows, 256 threads (8 waves). Wave w owns score rows {2w,2w+1}
// and output cols [16w,16w+16). Fixed per-row bound M_i kills all online
// rescaling: p = exp(s - M_i), lane-local partial sums, one end reduction.
// Double-buffered Pt tile -> single barrier per 64-column chunk.
__global__ void __launch_bounds__(256) gat_attn(
    const float* __restrict__ adj,
    const unsigned short* __restrict__ xpT,
    const float* __restrict__ s_src, const float* __restrict__ s_dst,
    const float* __restrict__ gmax,
    float* __restrict__ out)
{
    __shared__ __align__(16) unsigned short Pt[2][16][TJ];   // 2 x 2 KiB
    __shared__ __align__(16) float rowsum_s[16];

    const int tid  = threadIdx.x;
    const int lane = tid & 31;
    const int w    = tid >> 5;
    const int half = lane >> 4;
    const int l15  = lane & 15;
    const int i0   = blockIdx.x * 16;
    const int r0   = 2 * w, r1 = 2 * w + 1;

    const float g     = gmax[0];
    const float ssrc0 = s_src[i0 + r0];
    const float ssrc1 = s_src[i0 + r1];
    const float M0    = lrelu(ssrc0 + g);   // >= every valid score of row r0
    const float M1    = lrelu(ssrc1 + g);
    const float* arow0 = adj + (size_t)(i0 + r0) * N_NODES;
    const float* arow1 = adj + (size_t)(i0 + r1) * N_NODES;
    const unsigned short* bbase = xpT + (size_t)(w * 16 + l15) * N_NODES + half * 16;

    float acc0 = 0.f, acc1 = 0.f;           // lane-local partial row sums
    v8f c = {};

    // ---- pipeline prologue: loads for chunk 0 ----
    float av0 = __builtin_nontemporal_load(arow0 + lane);
    float av1 = __builtin_nontemporal_load(arow0 + lane + 32);
    float av2 = __builtin_nontemporal_load(arow1 + lane);
    float av3 = __builtin_nontemporal_load(arow1 + lane + 32);
    float sd0 = s_dst[lane];
    float sd1 = s_dst[lane + 32];
    ABu b0, b1;
    b0.q[0] = ((const uint4*)(bbase))[0];
    b0.q[1] = ((const uint4*)(bbase))[1];
    b1.q[0] = ((const uint4*)(bbase + 32))[0];
    b1.q[1] = ((const uint4*)(bbase + 32))[1];

    int buf = 0;
    for (int j0 = 0; j0 < N_NODES; j0 += TJ) {
        // ---- probabilities for rows r0,r1 (registers only) ----
        const float pA = prob(ssrc0 + sd0, av0, i0 + r0 == j0 + lane,      M0);
        const float pB = prob(ssrc0 + sd1, av1, i0 + r0 == j0 + lane + 32, M0);
        const float pC = prob(ssrc1 + sd0, av2, i0 + r1 == j0 + lane,      M1);
        const float pD = prob(ssrc1 + sd1, av3, i0 + r1 == j0 + lane + 32, M1);
        acc0 += pA + pB;
        acc1 += pC + pD;
        Pt[buf][r0][lane]      = f2bf(pA);
        Pt[buf][r0][lane + 32] = f2bf(pB);
        Pt[buf][r1][lane]      = f2bf(pC);
        Pt[buf][r1][lane + 32] = f2bf(pD);

        // ---- issue next chunk's loads; latency hides under barrier+WMMA ----
        const int jn = (j0 + TJ < N_NODES) ? (j0 + TJ) : 0;  // wrap: discarded
        const float nav0 = __builtin_nontemporal_load(arow0 + jn + lane);
        const float nav1 = __builtin_nontemporal_load(arow0 + jn + lane + 32);
        const float nav2 = __builtin_nontemporal_load(arow1 + jn + lane);
        const float nav3 = __builtin_nontemporal_load(arow1 + jn + lane + 32);
        const float nsd0 = s_dst[jn + lane];
        const float nsd1 = s_dst[jn + lane + 32];
        ABu nb0, nb1;
        nb0.q[0] = ((const uint4*)(bbase + jn))[0];
        nb0.q[1] = ((const uint4*)(bbase + jn))[1];
        nb1.q[0] = ((const uint4*)(bbase + jn + 32))[0];
        nb1.q[1] = ((const uint4*)(bbase + jn + 32))[1];

        __syncthreads();   // Pt[buf] ready; prior reads of Pt[buf^1] retired

        // ---- two K=32 WMMA accumulations: C += P @ x' ----
        const unsigned short (*pt)[TJ] = Pt[buf];
        ABu a;
        a.q[0] = *(const uint4*)&pt[l15][half * 16];
        a.q[1] = *(const uint4*)&pt[l15][half * 16 + 8];
        c = __builtin_amdgcn_wmma_f32_16x16x32_bf16(
                false, a.v, false, b0.v, (short)0, c, false, false);
        a.q[0] = *(const uint4*)&pt[l15][32 + half * 16];
        a.q[1] = *(const uint4*)&pt[l15][32 + half * 16 + 8];
        c = __builtin_amdgcn_wmma_f32_16x16x32_bf16(
                false, a.v, false, b1.v, (short)0, c, false, false);

        av0 = nav0; av1 = nav1; av2 = nav2; av3 = nav3;
        sd0 = nsd0; sd1 = nsd1;
        b0 = nb0;   b1 = nb1;
        buf ^= 1;
    }

    // ---- one end-of-row reduction, normalize, write h ----
    #pragma unroll
    for (int off = 16; off >= 1; off >>= 1) {
        acc0 += __shfl_xor(acc0, off, 32);
        acc1 += __shfl_xor(acc1, off, 32);
    }
    if (lane == 0) {
        rowsum_s[r0] = acc0;
        rowsum_s[r1] = acc1;
    }
    __syncthreads();
    #pragma unroll
    for (int v = 0; v < 8; ++v) {
        const int row = v + half * 8;
        const float inv = 1.f / fmaxf(rowsum_s[row], 1e-30f);
        out[(size_t)(i0 + row) * F_OUT + w * 16 + l15] = c[v] * inv;
    }
}

// ---------------------------------------------------------------------------
extern "C" void kernel_launch(void* const* d_in, const int* in_sizes, int n_in,
                              void* d_out, int out_size, void* d_ws, size_t ws_size,
                              hipStream_t stream) {
    const float* adj  = (const float*)d_in[0];   // 8192*8192
    const float* x    = (const float*)d_in[1];   // 8192*256
    const float* wgt  = (const float*)d_in[2];   // 256*128
    const float* bias = (const float*)d_in[3];   // 128
    const float* phi  = (const float*)d_in[4];   // 256
    float* out = (float*)d_out;                  // 8192*128

    char* ws = (char*)d_ws;
    unsigned short* xbf = (unsigned short*)(ws);                           // 4 MiB
    unsigned short* xpT = (unsigned short*)(ws + (size_t)4 * 1024 * 1024); // 2 MiB
    unsigned short* wT  = (unsigned short*)(ws + (size_t)6 * 1024 * 1024); // 64 KiB
    float* s_src = (float*)(ws + (size_t)6 * 1024 * 1024 + 65536);
    float* s_dst = (float*)(ws + (size_t)6 * 1024 * 1024 + 65536 + 32768);
    float* gmaxp = (float*)(ws + (size_t)6 * 1024 * 1024 + 65536 + 65536);

    gat_prep  <<<(N_NODES * F_IN) / 256, 256, 0, stream>>>(x, wgt, xbf, wT, s_src, s_dst);
    gat_xprime<<<N_NODES / 16,           256, 0, stream>>>(xbf, wT, bias, phi, xpT, s_src, s_dst);
    gat_gmax  <<<1,                      256, 0, stream>>>(s_dst, gmaxp);
    gat_attn  <<<N_NODES / 16,           256, 0, stream>>>(adj, xpT, s_src, s_dst, gmaxp, out);
}
